// Similarity_2534030705205
// MI455X (gfx1250) — matrix operand order repaired and checked by reading, not compiled, not run on hardware
//
#include <hip/hip_runtime.h>
#include <hip/hip_bf16.h>

typedef __bf16 bf16_t;
typedef __attribute__((ext_vector_type(16))) __bf16 v16bf;
typedef __attribute__((ext_vector_type(8)))  __bf16 v8bf;
typedef __attribute__((ext_vector_type(8)))  float  v8f;
typedef __attribute__((ext_vector_type(4)))  unsigned int u32x4;
typedef __attribute__((ext_vector_type(8)))  int i32x8;
typedef __attribute__((ext_vector_type(4)))  int i32x4;

#define BB 32
#define CC 64
#define HH 64
#define WW 64
#define II 128
#define KTOT 1152           // 2 planes * C * 9 positions, ordered slice*64 + c
#define KSTEP 32
#define KSTEPS (KTOT / KSTEP)   // 36
#define NTILE 128
#define APAD 40             // padded row length (halves): 80B pitch
#define ABUF_BYTES (II * APAD * 2)   // 10240 B per A buffer

#if __has_builtin(__builtin_amdgcn_tensor_load_to_lds)
#define USE_TDM 1
#else
#define USE_TDM 0
#endif

// ---------------------------------------------------------------------------
// Prep: build combined bf16 weight matrix Wc[I][KTOT] and t3[I].
//   k = slice*64 + c ; slice = (kh*3+kw)*2 + plane
//   plane 0 -> 2*|sw|*t   (multiplies x)
//   plane 1 -> -|sw|      (multiplies x^2)
//   t3[i] = sum |sw| * t^2
// ---------------------------------------------------------------------------
__global__ __launch_bounds__(128)
void sim_prep_kernel(const float* __restrict__ tmpl,
                     const float* __restrict__ sw,
                     bf16_t* __restrict__ Wc,
                     float* __restrict__ t3) {
    __shared__ float red[128];
    const int i = blockIdx.x;
    const int tid = threadIdx.x;

    float acc = 0.0f;
    for (int k = tid; k < KTOT; k += 128) {
        const int slice = k >> 6;
        const int c     = k & 63;
        const int plane = slice & 1;
        const int pos   = slice >> 1;
        const int kh    = pos / 3;
        const int kw    = pos % 3;
        const int idx4  = ((i * CC + c) * 3 + kh) * 3 + kw;
        const float wa  = fabsf(sw[idx4]);
        const float t   = tmpl[idx4];
        float wvv       = plane ? (-wa) : (2.0f * wa * t);
        Wc[i * KTOT + k] = (bf16_t)wvv;
        if (plane == 0) acc += wa * t * t;
    }
    red[tid] = acc;
    __syncthreads();
    for (int s = 64; s > 0; s >>= 1) {
        if (tid < s) red[tid] += red[tid + s];
        __syncthreads();
    }
    if (tid == 0) t3[i] = red[0];
}

// ---------------------------------------------------------------------------
// Main implicit-GEMM kernel. out = Wc @ im2col([x, x^2]) - t3
// 256 threads (8 wave32). Tile: M=128 x N=128 pixels.
// Double-buffered pipeline, one barrier per k-step:
//   issue TDM(ks+1) + global loads(ks+1) -> WMMA(ks) -> cvt/ds_store(ks+1)
//   -> s_wait_tensorcnt -> barrier
// ---------------------------------------------------------------------------
__global__ __launch_bounds__(256)
void sim_wmma_kernel(const float* __restrict__ x,
                     const bf16_t* __restrict__ Wc,
                     const float* __restrict__ t3,
                     float* __restrict__ out) {
    __shared__ bf16_t As[2][II][APAD];      // 2 x (128 rows x 32 k, 80B pitch)
    __shared__ bf16_t Bs[2][NTILE][APAD];   // 2 x (128 pixels x 32 k)
    __shared__ float  t3s[II];

    const int tid  = threadIdx.x;
    const int lane = tid & 31;
    const int wv   = tid >> 5;
    const int m0   = (wv & 1) * 64;
    const int n0   = (wv >> 1) * 32;
    const int tileBase = blockIdx.x * NTILE;

    if (tid < II) t3s[tid] = t3[tid];

    const int halfSel = (lane < 16) ? 0 : 1;
    const int rowLane = lane & 15;

    v8f acc[4][2];
#pragma unroll
    for (int mt = 0; mt < 4; ++mt)
#pragma unroll
        for (int nt = 0; nt < 2; ++nt)
            acc[mt][nt] = (v8f){0.f, 0.f, 0.f, 0.f, 0.f, 0.f, 0.f, 0.f};

    // ---- B staging: this thread owns one pixel column, 16 channels ----
    const int bN     = tid & 127;        // pixel within tile
    const int bCHalf = tid >> 7;         // 0/1 -> channels [0,16) or [16,32)
    const int bP  = tileBase + bN;
    const int bB  = bP >> 12;            // image index
    const int bHW = bP & 4095;
    const int bH  = bHW >> 6;
    const int bW  = bHW & 63;

    // Load next k-step's 16 channel values into registers (predicate hoisted).
    auto loadStage = [&](int ksn, float* regs) -> int {
        const int k0    = ksn * KSTEP;
        const int slice = k0 >> 6;
        const int plane = slice & 1;
        const int pos   = slice >> 1;
        const int kh    = pos / 3;
        const int kw    = pos % 3;
        const int cBase = k0 & 63;
        const int hs    = bH + kh - 1;
        const int wsrc  = bW + kw - 1;
        const bool valid = ((unsigned)hs < (unsigned)HH) &&
                           ((unsigned)wsrc < (unsigned)WW);
        const float* src = x + ((size_t)(bB * CC + cBase + bCHalf * 16)) * (HH * WW)
                             + hs * WW + wsrc;
        if (cBase == 0) __builtin_prefetch(src + 32 * HH * WW, 0, 1);
#pragma unroll
        for (int c = 0; c < 16; ++c)
            regs[c] = valid ? src[c * (HH * WW)] : 0.0f;
        return plane;
    };
    // Square (plane 1), convert to bf16, and store into LDS buffer `buf`.
    auto storeStage = [&](int buf, int plane, const float* regs) {
        bf16_t vals[16];
#pragma unroll
        for (int c = 0; c < 16; ++c) {
            float v = regs[c];
            if (plane) v *= v;
            vals[c] = (bf16_t)v;
        }
        *(v8bf*)(&Bs[buf][bN][bCHalf * 16])     = *(const v8bf*)&vals[0];
        *(v8bf*)(&Bs[buf][bN][bCHalf * 16 + 8]) = *(const v8bf*)&vals[8];
    };

#if USE_TDM
    // ---- TDM descriptor (uniform): 2D tile 32 cols x 128 rows of bf16 ----
    const unsigned ldsA = (unsigned)(uintptr_t)(&As[0][0][0]);
    const unsigned long long wcAddr = (unsigned long long)(uintptr_t)Wc;
    i32x8 g1;
    g1[0] = (1 << 16)        // data_size = 1 -> 2 bytes
          | (1 << 20)        // pad_enable
          | (3 << 22)        // pad_interval code 3 -> every 16 dwords (64B row)
          | (3 << 25);       // pad_amount code 3 -> 4 dwords (16B) pad
    g1[1] = (KTOT & 0xffff) << 16;                 // tensor_dim0[15:0]
    g1[2] = (KTOT >> 16) | ((II & 0xffff) << 16);  // tensor_dim0[31:16] | tensor_dim1[15:0]
    g1[3] = (II >> 16) | (KSTEP << 16);            // tensor_dim1[31:16] | tile_dim0
    g1[4] = NTILE & 0xffff;                        // tile_dim1 = 128, tile_dim2 = 0
    g1[5] = KTOT;                                  // tensor_dim0_stride[31:0]
    g1[6] = 0;
    g1[7] = 0;
    const i32x4 g2 = (i32x4){0, 0, 0, 0};
    const i32x4 g3 = (i32x4){0, 0, 0, 0};
    const i32x8 g4 = (i32x8){0, 0, 0, 0, 0, 0, 0, 0};

    auto issueTDM = [&](int ksn, int buf) {
        const unsigned long long ga = wcAddr + (unsigned long long)(ksn * KSTEP) * 2ull;
        u32x4 g0;
        g0[0] = 1u;                                   // count=1 user D#
        g0[1] = ldsA + (unsigned)buf * ABUF_BYTES;    // lds_addr
        g0[2] = (unsigned)(ga & 0xffffffffull);       // global_addr[31:0]
        g0[3] = (unsigned)((ga >> 32) & 0x1ffffffu)   // global_addr[56:32]
              | (2u << 30);                           // type = 2 (image)
        __builtin_amdgcn_tensor_load_to_lds(g0, g1, g2, g3, g4, 0);
    };
#else
    auto issueManualA = [&](int ksn, int buf) {
        const int aRow = tid >> 1;
        const int aKK  = (tid & 1) * 16;
        const bf16_t* src = Wc + aRow * KTOT + ksn * KSTEP + aKK;
        v8bf a0 = *(const v8bf*)(src);
        v8bf a1 = *(const v8bf*)(src + 8);
        *(v8bf*)(&As[buf][aRow][aKK])     = a0;
        *(v8bf*)(&As[buf][aRow][aKK + 8]) = a1;
    };
#endif

    // ---------------- prologue: stage k-step 0 into buffer 0 ----------------
    {
        float stage[16];
        const int plane0 = loadStage(0, stage);
#if USE_TDM
        if (wv == 0) issueTDM(0, 0);
#else
        issueManualA(0, 0);
#endif
        storeStage(0, plane0, stage);
#if USE_TDM
        if (wv == 0) __builtin_amdgcn_s_wait_tensorcnt(0);
#endif
    }
    __syncthreads();

    // ---------------- main pipelined loop: one barrier per k-step ----------
#pragma unroll 1
    for (int ks = 0; ks < KSTEPS; ++ks) {
        const int cur = ks & 1;
        const int nxt = cur ^ 1;
        const bool hasNext = (ks + 1) < KSTEPS;

        float stage[16];
        int planeN = 0;
        if (hasNext) {
#if USE_TDM
            if (wv == 0) issueTDM(ks + 1, nxt);    // async A panel -> LDS
#else
            issueManualA(ks + 1, nxt);
#endif
            planeN = loadStage(ks + 1, stage);     // global loads in flight
        }

        // ---- WMMA on current buffers (overlaps the staging traffic) ----
        v16bf bfrag[2];
#pragma unroll
        for (int nt = 0; nt < 2; ++nt) {
            const bf16_t* bp = &Bs[cur][n0 + nt * 16 + rowLane][halfSel * 8];
            v8bf b0 = *(const v8bf*)(bp);
            v8bf b1 = *(const v8bf*)(bp + 16);
            bfrag[nt] = __builtin_shufflevector(b0, b1,
                0, 1, 2, 3, 4, 5, 6, 7, 8, 9, 10, 11, 12, 13, 14, 15);
        }
#pragma unroll
        for (int mt = 0; mt < 4; ++mt) {
            const bf16_t* ap = &As[cur][m0 + mt * 16 + rowLane][halfSel * 8];
            v8bf aLo = *(const v8bf*)(ap);        // K base .. base+7
            v8bf aHi = *(const v8bf*)(ap + 16);   // K base+16 .. base+23
            v16bf afrag = __builtin_shufflevector(aLo, aHi,
                0, 1, 2, 3, 4, 5, 6, 7, 8, 9, 10, 11, 12, 13, 14, 15);
#pragma unroll
            for (int nt = 0; nt < 2; ++nt) {
                acc[mt][nt] = __builtin_amdgcn_wmma_f32_16x16x32_bf16(
                    false, afrag, false, bfrag[nt],
                    (short)0, acc[mt][nt], false, false);
            }
        }

        if (hasNext) {
            storeStage(nxt, planeN, stage);        // cvt + 2x ds_store_b128
#if USE_TDM
            if (wv == 0) __builtin_amdgcn_s_wait_tensorcnt(0);
#endif
        }
        __syncthreads();
    }

    // ---- epilogue: out[b,i,h,w] = acc - t3[i] ----
#pragma unroll
    for (int mt = 0; mt < 4; ++mt) {
#pragma unroll
        for (int nt = 0; nt < 2; ++nt) {
            const int n  = n0 + nt * 16 + rowLane;
            const int p  = tileBase + n;
            const int b  = p >> 12;
            const int hw = p & 4095;
#pragma unroll
            for (int v = 0; v < 8; ++v) {
                const int i = m0 + mt * 16 + halfSel * 8 + v;
                out[(b * II + i) * (HH * WW) + hw] = acc[mt][nt][v] - t3s[i];
            }
        }
    }
}

// ---------------------------------------------------------------------------
extern "C" void kernel_launch(void* const* d_in, const int* in_sizes, int n_in,
                              void* d_out, int out_size, void* d_ws, size_t ws_size,
                              hipStream_t stream) {
    const float* x    = (const float*)d_in[0];   // [B,C,H,W]
    const float* tmpl = (const float*)d_in[1];   // [I,C,3,3]
    const float* sw   = (const float*)d_in[2];   // [I,C,3,3]
    float* out        = (float*)d_out;           // [B,I,H,W]

    bf16_t* Wc = (bf16_t*)d_ws;                              // 128*1152*2 B
    float*  t3 = (float*)((char*)d_ws + (size_t)II * KTOT * sizeof(bf16_t));

    sim_prep_kernel<<<dim3(II), dim3(128), 0, stream>>>(tmpl, sw, Wc, t3);

    const int nTiles = (BB * HH * WW) / NTILE;   // 1024
    sim_wmma_kernel<<<dim3(nTiles), dim3(256), 0, stream>>>(x, Wc, t3, out);
}
